// InputAttentionEncoder_scaled_dot_product_26680336843490
// MI455X (gfx1250) — matrix operand order repaired and checked by reading, not compiled, not run on hardware
//
#include <hip/hip_runtime.h>

// Problem constants (match reference).
#define B_  128
#define T_  256
#define N_  512
#define M_  512
#define FM  (4 * M_)      // 2048
#define K_  512           // shared K for both GEMMs (N_ == M_)

typedef __attribute__((ext_vector_type(16))) __bf16 v16bf;
typedef __attribute__((ext_vector_type(8)))  float  v8f;

union BfFrag {
    v16bf v;
    uint4 q[2];   // 2 x 16B = 16 bf16
};

// float -> bf16 bits, round-to-nearest-even.
__device__ __forceinline__ unsigned short f2bf(float f) {
    unsigned int u = __float_as_uint(f);
    unsigned int r = u + 0x7FFFu + ((u >> 16) & 1u);
    return (unsigned short)(r >> 16);
}

__device__ __forceinline__ float sigmoidf_(float x) {
    return 1.0f / (1.0f + __expf(-x));
}

// ---------------------------------------------------------------------------
// Prep: bf16 weights (w_ih pre-scaled by 1/N == exact alpha), fused bias.
// ---------------------------------------------------------------------------
__global__ void prep_kernel(const float* __restrict__ w_ih,
                            const float* __restrict__ w_hh,
                            const float* __restrict__ b_ih,
                            const float* __restrict__ b_hh,
                            unsigned short* __restrict__ w_ih_bf,
                            unsigned short* __restrict__ w_hh_bf,
                            float* __restrict__ bias) {
    int i = blockIdx.x * blockDim.x + threadIdx.x;
    const int WN = FM * K_;                       // 1,048,576 per weight
    if (i < WN) {
        w_ih_bf[i] = f2bf(w_ih[i] * (1.0f / (float)N_));   // fold softmax(const)=1/N
    } else if (i < 2 * WN) {
        int j = i - WN;
        w_hh_bf[j] = f2bf(w_hh[j]);
    } else if (i < 2 * WN + FM) {
        int j = i - 2 * WN;
        bias[j] = b_ih[j] + b_hh[j];
    }
}

// Vectorized f32 -> bf16: 4 floats per thread (float4 in, uint2 out).
__global__ void cvt_x_kernel(const float4* __restrict__ x,
                             uint2* __restrict__ xbf, int n4) {
    int i = blockIdx.x * blockDim.x + threadIdx.x;
    if (i < n4) {
        float4 f = x[i];
        uint2 o;
        o.x = (unsigned)f2bf(f.x) | ((unsigned)f2bf(f.y) << 16);
        o.y = (unsigned)f2bf(f.z) | ((unsigned)f2bf(f.w) << 16);
        xbf[i] = o;
    }
}

__global__ void zero_state_kernel(unsigned short* __restrict__ hbf,
                                  float* __restrict__ c) {
    int i = blockIdx.x * blockDim.x + threadIdx.x;
    if (i < B_ * M_) { hbf[i] = 0; c[i] = 0.0f; }
}

// ---------------------------------------------------------------------------
// 2x2 register-blocked WMMA: one wave computes a 32x32 output region
// (4 x 16x16 tiles), reusing each A/B fragment twice -> 2 loads per WMMA.
// A: row-major (rows, K_) bf16.  W: row-major (FM, K_) bf16 (used as B^T).
// Fragment layouts per CDNA5 ISA 7.12.2 (16-bit A 16x32, B 32x16, C 16x16).
// ---------------------------------------------------------------------------
__device__ __forceinline__ void wmma_tile2x2(const unsigned short* __restrict__ A,
                                             const unsigned short* __restrict__ W,
                                             float* __restrict__ C,
                                             const float* __restrict__ bias,
                                             int rowTile32, int colTile32) {
    const int lane = threadIdx.x;        // wave32 lane
    const int half = (lane >> 4) & 1;    // 0: lanes 0-15, 1: lanes 16-31
    const int l15  = lane & 15;

    const unsigned short* arow0 = A + (size_t)(rowTile32 * 32 + l15) * K_;
    const unsigned short* arow1 = arow0 + (size_t)16 * K_;
    const unsigned short* wrow0 = W + (size_t)(colTile32 * 32 + l15) * K_;
    const unsigned short* wrow1 = wrow0 + (size_t)16 * K_;
    const int ka = half * 8;             // A: K offsets {0..7,16..23} / {8..15,24..31}
    const int kb = half * 8;             // B: K offsets {0..7,16..23} / {8..15,24..31}

    v8f acc00 = {}, acc01 = {}, acc10 = {}, acc11 = {};
#pragma unroll
    for (int k = 0; k < K_; k += 32) {
        BfFrag a0, a1, b0, b1;
        a0.q[0] = *(const uint4*)(arow0 + k + ka);
        a0.q[1] = *(const uint4*)(arow0 + k + ka + 16);
        a1.q[0] = *(const uint4*)(arow1 + k + ka);
        a1.q[1] = *(const uint4*)(arow1 + k + ka + 16);
        b0.q[0] = *(const uint4*)(wrow0 + k + kb);
        b0.q[1] = *(const uint4*)(wrow0 + k + kb + 16);
        b1.q[0] = *(const uint4*)(wrow1 + k + kb);
        b1.q[1] = *(const uint4*)(wrow1 + k + kb + 16);
        acc00 = __builtin_amdgcn_wmma_f32_16x16x32_bf16(false, a0.v, false, b0.v,
                                                        (short)0, acc00, false, false);
        acc01 = __builtin_amdgcn_wmma_f32_16x16x32_bf16(false, a0.v, false, b1.v,
                                                        (short)0, acc01, false, false);
        acc10 = __builtin_amdgcn_wmma_f32_16x16x32_bf16(false, a1.v, false, b0.v,
                                                        (short)0, acc10, false, false);
        acc11 = __builtin_amdgcn_wmma_f32_16x16x32_bf16(false, a1.v, false, b1.v,
                                                        (short)0, acc11, false, false);
    }

    const int col0 = colTile32 * 32 + l15;
    const int col1 = col0 + 16;
    const float bv0 = bias ? bias[col0] : 0.0f;
    const float bv1 = bias ? bias[col1] : 0.0f;
    const int row0 = rowTile32 * 32 + half * 8;   // C: VGPR r -> M=r (+8 upper half)
    const int row1 = row0 + 16;
#pragma unroll
    for (int r = 0; r < 8; ++r) {
        C[(size_t)(row0 + r) * FM + col0] = acc00[r] + bv0;
        C[(size_t)(row0 + r) * FM + col1] = acc01[r] + bv1;
        C[(size_t)(row1 + r) * FM + col0] = acc10[r] + bv0;
        C[(size_t)(row1 + r) * FM + col1] = acc11[r] + bv1;
    }
}

// Gx(B*T, 4M) = xs_bf(B*T, N) @ w_ih_bf^T + bias   — fully parallel, hoisted.
__global__ void gemm_x_kernel(const unsigned short* __restrict__ xbf,
                              const unsigned short* __restrict__ wihbf,
                              const float* __restrict__ bias,
                              float* __restrict__ Gx) {
    const int rowTile32 = blockIdx.y * blockDim.y + threadIdx.y;  // 0..1023
    const int colTile32 = blockIdx.x;                             // 0..63
    wmma_tile2x2(xbf, wihbf, Gx, bias, rowTile32, colTile32);
}

// Gh(B, 4M) = h_bf(B, M) @ w_hh_bf^T   — one launch per LSTM step.
__global__ void gemm_h_kernel(const unsigned short* __restrict__ hbf,
                              const unsigned short* __restrict__ whhbf,
                              float* __restrict__ Gh) {
    const int rowTile32 = threadIdx.y;   // 0..3  (B/32)
    const int colTile32 = blockIdx.x;    // 0..63
    wmma_tile2x2(hbf, whhbf, Gh, nullptr, rowTile32, colTile32);
}

// Fused LSTM cell: gates -> (i,f,g,o) -> c,h; emits f32 h to out and bf16 h
// for the next step's GEMM.
__global__ void lstm_cell_kernel(const float* __restrict__ Gx,
                                 const float* __restrict__ Gh,
                                 float* __restrict__ c,
                                 unsigned short* __restrict__ hbf,
                                 float* __restrict__ out, int t) {
    const int idx = blockIdx.x * blockDim.x + threadIdx.x;  // 0..B*M-1
    const int b = idx >> 9;        // / M_
    const int m = idx & (M_ - 1);
    const float* gx = Gx + (size_t)(b * T_ + t) * FM;
    const float* gh = Gh + (size_t)b * FM;

    const float il = gx[m]          + gh[m];
    const float fl = gx[M_ + m]     + gh[M_ + m];
    const float gl = gx[2 * M_ + m] + gh[2 * M_ + m];
    const float ol = gx[3 * M_ + m] + gh[3 * M_ + m];

    const float ig = sigmoidf_(il);
    const float fg = sigmoidf_(fl);
    const float gg = tanhf(gl);
    const float og = sigmoidf_(ol);

    const float cn = fg * c[idx] + ig * gg;
    c[idx] = cn;
    const float h = og * tanhf(cn);

    out[((size_t)b * T_ + t) * M_ + m] = h;   // output layout (B, T, M)
    hbf[idx] = f2bf(h);
}

// ---------------------------------------------------------------------------
extern "C" void kernel_launch(void* const* d_in, const int* in_sizes, int n_in,
                              void* d_out, int out_size, void* d_ws, size_t ws_size,
                              hipStream_t stream) {
    const float* inputs = (const float*)d_in[0];
    // d_in[1..3] (W_e_w, W_e_b, U_e_w) are dead: softmax of a row-constant is
    // exactly 1/N, so alpha == 1/512 regardless of the attention branch.
    const float* w_ih = (const float*)d_in[4];
    const float* w_hh = (const float*)d_in[5];
    const float* b_ih = (const float*)d_in[6];
    const float* b_hh = (const float*)d_in[7];
    float* out = (float*)d_out;

    char* p = (char*)d_ws;
    auto alloc = [&](size_t bytes) -> char* {
        char* r = p;
        p += (bytes + 255) & ~(size_t)255;
        return r;
    };
    unsigned short* xbf   = (unsigned short*)alloc((size_t)B_ * T_ * N_ * 2); // 33.5 MB
    unsigned short* wihbf = (unsigned short*)alloc((size_t)FM * K_ * 2);      //  2 MB
    unsigned short* whhbf = (unsigned short*)alloc((size_t)FM * K_ * 2);      //  2 MB
    float*          bias  = (float*)alloc((size_t)FM * 4);
    float*          Gx    = (float*)alloc((size_t)B_ * T_ * FM * 4);          // 268 MB
    float*          Gh    = (float*)alloc((size_t)B_ * FM * 4);               //  1 MB
    unsigned short* hbf   = (unsigned short*)alloc((size_t)B_ * M_ * 2);
    float*          cst   = (float*)alloc((size_t)B_ * M_ * 4);

    {   // weight conversion + bias fusion
        const int total = 2 * FM * K_ + FM;
        prep_kernel<<<(total + 255) / 256, 256, 0, stream>>>(
            w_ih, w_hh, b_ih, b_hh, wihbf, whhbf, bias);
    }
    {   // input conversion (x4 vectorized)
        const int n4 = (B_ * T_ * N_) / 4;
        cvt_x_kernel<<<(n4 + 255) / 256, 256, 0, stream>>>(
            (const float4*)inputs, (uint2*)xbf, n4);
    }
    zero_state_kernel<<<(B_ * M_ + 255) / 256, 256, 0, stream>>>(hbf, cst);

    {   // hoisted input-projection GEMM:
        // 1024 macro-row tiles (32 rows) x 64 macro-col tiles (32 cols)
        dim3 grid(FM / 32, (B_ * T_ / 32) / 8);
        dim3 blk(32, 8);
        gemm_x_kernel<<<grid, blk, 0, stream>>>(xbf, wihbf, bias, Gx);
    }

    // Sequential recurrence: 256 (GEMM, cell) pairs, graph-friendly.
    for (int t = 0; t < T_; ++t) {
        gemm_h_kernel<<<dim3(FM / 32), dim3(32, 4), 0, stream>>>(hbf, whhbf, Gh);
        lstm_cell_kernel<<<(B_ * M_) / 256, 256, 0, stream>>>(Gx, Gh, cst, hbf, out, t);
    }
}
